// MatchingNetwork_33019708572203
// MI455X (gfx1250) — compile-verified, matrix operand order checked by reading, not run on hardware
//
#include <hip/hip_runtime.h>
#include <hip/hip_bf16.h>
#include <math.h>

// Problem constants
constexpr int kE    = 64;     // embedding / hidden dim
constexpr int kG    = 256;    // 4*E gate width
constexpr int kSEQ  = 200;    // N*k support sequence length
constexpr int kSEQP = 224;    // padded seq (multiple of 32 for K-loops)
constexpr int kB    = 64;     // batch
constexpr int kT    = 128;    // target length
constexpr int kLS   = 40;     // tokens per item
constexpr int kRT   = kB * kT;    // 8192 target rows
constexpr int kRS   = kB * kSEQ;  // 12800 support rows

typedef __attribute__((ext_vector_type(16))) _Float16 v16h;
typedef __attribute__((ext_vector_type(8)))  float    v8f;

static __device__ inline v8f wmma16(v16h a, v16h b, v8f c) {
  return __builtin_amdgcn_wmma_f32_16x16x32_f16(
      /*neg_a=*/false, a, /*neg_b=*/false, b,
      /*c_mod=*/(short)0, c, /*reuse_a=*/false, /*reuse_b=*/false);
}

static __device__ inline int kmap(int e) { return (e < 8) ? e : (e + 8); }
static __device__ inline float sigm(float x) { return 1.0f / (1.0f + __expf(-x)); }

// ---- fragment loaders (ISA 7.12.2 layouts, wave32) ----
// All loaders are branch-free: callers guarantee in-range via exact sizes or
// zero-padded staging buffers. Per-lane reads are two contiguous 8-element
// runs -> vectorizes to b128 loads + v_cvt_pk_f16_f32.

// A: 16x32 tile from row-major f32 (global or generic), no bounds
static __device__ inline v16h fragA_f32(const float* __restrict__ src, int ld,
                                        int row0, int k0, int lane) {
  v16h a;
  int r = row0 + (lane & 15);
  int kb = k0 + ((lane >> 4) << 3);
  const float* p = src + (size_t)r * ld + kb;
  #pragma unroll
  for (int e = 0; e < 16; ++e) a[e] = (_Float16)p[kmap(e)];
  return a;
}

// A: 16x32 tile from row-major f16 LDS
static __device__ inline v16h fragA_lds16(const _Float16* __restrict__ src, int ld,
                                          int row0, int k0, int lane) {
  v16h a;
  int r = row0 + (lane & 15);
  int kb = k0 + ((lane >> 4) << 3);
  const _Float16* p = src + (size_t)r * ld + kb;
  #pragma unroll
  for (int e = 0; e < 16; ++e) a[e] = p[kmap(e)];
  return a;
}

// A: 16x32 tile from row-major f32 LDS
static __device__ inline v16h fragA_ldsf32(const float* __restrict__ src, int ld,
                                           int row0, int k0, int lane) {
  v16h a;
  int r = row0 + (lane & 15);
  int kb = k0 + ((lane >> 4) << 3);
  const float* p = src + (size_t)r * ld + kb;
  #pragma unroll
  for (int e = 0; e < 16; ++e) a[e] = (_Float16)p[kmap(e)];
  return a;
}

// B = W^T where W is row-major (N,K) f32 global: B(k,n) = W[n*ld + k]
static __device__ inline v16h fragBT_f32(const float* __restrict__ W, int ld,
                                         int k0, int n0, int lane) {
  v16h b;
  int n = n0 + (lane & 15);
  int kb = k0 + ((lane >> 4) << 3);
  const float* p = W + (size_t)n * ld + kb;
  #pragma unroll
  for (int e = 0; e < 16; ++e) b[e] = (_Float16)p[kmap(e)];
  return b;
}

// B = S^T from row-major (N,K) f16 LDS
static __device__ inline v16h fragBT_lds16(const _Float16* __restrict__ S, int ld,
                                           int k0, int n0, int lane) {
  v16h b;
  int n = n0 + (lane & 15);
  int kb = k0 + ((lane >> 4) << 3);
  const _Float16* p = S + (size_t)n * ld + kb;
  #pragma unroll
  for (int e = 0; e < 16; ++e) b[e] = p[kmap(e)];
  return b;
}

// B (non-transposed): row-major (K,N) f16 LDS
static __device__ inline v16h fragB_lds16(const _Float16* __restrict__ S, int ld,
                                          int k0, int n0, int lane) {
  v16h b;
  int n = n0 + (lane & 15);
  int kb = k0 + ((lane >> 4) << 3);
  #pragma unroll
  for (int e = 0; e < 16; ++e) b[e] = S[(size_t)(kb + kmap(e)) * ld + n];
  return b;
}

// C/D stores (no guards; all dims exact multiples of 16)
static __device__ inline void storeC_g(float* __restrict__ dst, int ld, int row0,
                                       int col0, v8f c, int lane) {
  int col = col0 + (lane & 15);
  int rb = row0 + ((lane >> 4) << 3);
  #pragma unroll
  for (int v = 0; v < 8; ++v) dst[(size_t)(rb + v) * ld + col] = c[v];
}

static __device__ inline void storeC_lds16(_Float16* __restrict__ dst, int ld,
                                           int row0, int col0, v8f c, int lane) {
  int col = col0 + (lane & 15);
  int rb = row0 + ((lane >> 4) << 3);
  #pragma unroll
  for (int v = 0; v < 8; ++v) dst[(size_t)(rb + v) * ld + col] = (_Float16)c[v];
}

static __device__ inline void storeC_ldsf32(float* __restrict__ dst, int ld,
                                            int row0, int col0, v8f c, int lane) {
  int col = col0 + (lane & 15);
  int rb = row0 + ((lane >> 4) << 3);
  #pragma unroll
  for (int v = 0; v < 8; ++v) dst[(size_t)(rb + v) * ld + col] = c[v];
}

// =================== kernels ===================

// out[row][e] = sum_l emb[tok[row][l]][e]   (one 64-thread block per row)
__global__ void k_embed_sum(const int* __restrict__ tok, const float* __restrict__ emb,
                            float* __restrict__ out, int rows, int L) {
  int row = blockIdx.x;
  if (row >= rows) return;
  int e = threadIdx.x;
  const int* tr = tok + (size_t)row * L;
  float s = 0.0f;
  for (int l = 0; l < L; ++l) s += emb[(size_t)tr[l] * kE + e];
  out[(size_t)row * kE + e] = s;
}

// Y(M,256) = X(M,64) @ W(256,64)^T + b0 + b1    (8 waves, 16x128 per block-y half)
__global__ void k_gemm_xw(const float* __restrict__ X, const float* __restrict__ W,
                          const float* __restrict__ b0, const float* __restrict__ b1,
                          float* __restrict__ Y) {
  int lane = threadIdx.x & 31, wave = threadIdx.x >> 5;
  int row0 = blockIdx.x * 16;
  int col0 = blockIdx.y * 128 + wave * 16;
  v8f c = {};
  #pragma unroll
  for (int k0 = 0; k0 < kE; k0 += 32) {
    v16h a = fragA_f32(X, kE, row0, k0, lane);
    v16h b = fragBT_f32(W, kE, k0, col0, lane);
    c = wmma16(a, b, c);
  }
  int col = col0 + (lane & 15);
  int rb = row0 + ((lane >> 4) << 3);
  float bias = b0[col] + b1[col];
  #pragma unroll
  for (int v = 0; v < 8; ++v) Y[(size_t)(rb + v) * kG + col] = c[v] + bias;
}

// Z(M,256) = H(M,64) @ W(256,64)^T + XZ(M,256)
__global__ void k_cell_z(const float* __restrict__ H, const float* __restrict__ W,
                         const float* __restrict__ XZ, float* __restrict__ Z) {
  int lane = threadIdx.x & 31, wave = threadIdx.x >> 5;
  int row0 = blockIdx.x * 16;
  int col0 = blockIdx.y * 128 + wave * 16;
  v8f c = {};
  #pragma unroll
  for (int k0 = 0; k0 < kE; k0 += 32) {
    v16h a = fragA_f32(H, kE, row0, k0, lane);
    v16h b = fragBT_f32(W, kE, k0, col0, lane);
    c = wmma16(a, b, c);
  }
  int col = col0 + (lane & 15);
  int rb = row0 + ((lane >> 4) << 3);
  #pragma unroll
  for (int v = 0; v < 8; ++v)
    Z[(size_t)(rb + v) * kG + col] = c[v] + XZ[(size_t)(rb + v) * kG + col];
}

// Persistent BiLSTM recurrence. 4 blocks: blockIdx.x = {half | dir<<1}, 512 thr = 16 waves.
// Per step: z(32,256) = h(32,64) @ Whh^T via WMMA from LDS, gates elementwise, h back to LDS.
__global__ void k_bilstm(const float* __restrict__ xz_f, const float* __restrict__ xz_b,
                         const float* __restrict__ Wf, const float* __restrict__ Wb,
                         float* __restrict__ hf, float* __restrict__ hb) {
  __shared__ _Float16 sW[kG * kE];   // 32 KB: Whh as f16
  __shared__ _Float16 sh[32 * kE];   // 4 KB:  h (batch-half x E)
  __shared__ _Float16 sz[32 * kG];   // 16 KB: z staging
  int dir = blockIdx.x >> 1;
  int half = blockIdx.x & 1;
  const float* xz = dir ? xz_b : xz_f;
  const float* W  = dir ? Wb : Wf;
  float* hout     = dir ? hb : hf;
  int lane = threadIdx.x & 31, wave = threadIdx.x >> 5;

  for (int i = threadIdx.x; i < kG * kE; i += 512) sW[i] = (_Float16)W[i];
  for (int i = threadIdx.x; i < 32 * kE; i += 512) sh[i] = (_Float16)0.0f;
  __syncthreads();

  float creg[4] = {0.0f, 0.0f, 0.0f, 0.0f};

  for (int t = 0; t < kSEQ; ++t) {
    int ti = dir ? (kSEQ - 1 - t) : t;
    // z = h @ Whh^T : 32x256, 32 tiles, 2 per wave
    #pragma unroll
    for (int j = 0; j < 2; ++j) {
      int tid2 = wave * 2 + j;
      int r0 = (tid2 & 1) << 4;
      int c0 = (tid2 >> 1) << 4;
      v8f acc = {};
      acc = wmma16(fragA_lds16(sh, kE, r0, 0, lane),  fragBT_lds16(sW, kE, 0, c0, lane),  acc);
      acc = wmma16(fragA_lds16(sh, kE, r0, 32, lane), fragBT_lds16(sW, kE, 32, c0, lane), acc);
      storeC_lds16(sz, kG, r0, c0, acc, lane);
    }
    __syncthreads();
    // gates: 32*64 cells, 4 per thread
    #pragma unroll
    for (int i = 0; i < 4; ++i) {
      int id = threadIdx.x + (i << 9);
      int row = id >> 6, col = id & 63;
      int bat = (half << 5) + row;
      const float* xzr = xz + ((size_t)bat * kSEQ + ti) * kG;
      float zi = (float)sz[row * kG + col      ] + xzr[col      ];
      float zf = (float)sz[row * kG + col +  64] + xzr[col +  64];
      float zg = (float)sz[row * kG + col + 128] + xzr[col + 128];
      float zo = (float)sz[row * kG + col + 192] + xzr[col + 192];
      float cn = sigm(zf) * creg[i] + sigm(zi) * tanhf(zg);
      creg[i] = cn;
      float hv = sigm(zo) * tanhf(cn);
      sh[row * kE + col] = (_Float16)hv;
      hout[((size_t)bat * kSEQ + ti) * kE + col] = hv;
    }
    __syncthreads();
  }
}

// sup_emb = hf + hb + enc ; sn = ||sup_emb||  (64-thread block per row)
__global__ void k_sup_combine(const float* __restrict__ hf, const float* __restrict__ hb,
                              const float* __restrict__ enc, float* __restrict__ semb,
                              float* __restrict__ sn) {
  __shared__ float red[64];
  int row = blockIdx.x;
  int e = threadIdx.x;
  size_t idx = (size_t)row * kE + e;
  float v = hf[idx] + hb[idx] + enc[idx];
  semb[idx] = v;
  red[e] = v * v;
  __syncthreads();
  for (int s2 = 32; s2 > 0; s2 >>= 1) {
    if (e < s2) red[e] += red[e + s2];
    __syncthreads();
  }
  if (e == 0) sn[row] = sqrtf(red[0]);
}

__global__ void k_zero3(float* a, float* b, float* c, int n) {
  int i = blockIdx.x * blockDim.x + threadIdx.x;
  if (i < n) { a[i] = 0.0f; b[i] = 0.0f; c[i] = 0.0f; }
}

__global__ void k_add_inplace(float* __restrict__ h, const float* __restrict__ r, int n) {
  int i = blockIdx.x * blockDim.x + threadIdx.x;
  if (i < n) h[i] += r[i];
}

// gates: c' = sig(f)c + sig(i)tanh(g); hn = sig(o)tanh(c') + x
__global__ void k_gates(const float* __restrict__ Z, const float* __restrict__ X,
                        float* __restrict__ C, float* __restrict__ Hn, int M) {
  int i = blockIdx.x * blockDim.x + threadIdx.x;
  if (i >= M * kE) return;
  int row = i >> 6, col = i & 63;
  const float* zr = Z + (size_t)row * kG;
  float zi = zr[col], zf = zr[col + 64], zg = zr[col + 128], zo = zr[col + 192];
  float cn = sigm(zf) * C[i] + sigm(zi) * tanhf(zg);
  C[i] = cn;
  Hn[i] = sigm(zo) * tanhf(cn) + X[i];
}

// Fused scores -> softmax -> readout. Block = (b, 16 target rows), 256 thr = 8 waves.
// sS padded to 224 rows (zeros), sAtt padded to 224 cols (zeros) => branch-free WMMA.
__global__ void k_att(const float* __restrict__ hA, const float* __restrict__ sup,
                      float* __restrict__ rout) {
  __shared__ _Float16 sS[kSEQP * kE];  // 28 KB support embeds f16, rows 200..223 = 0
  __shared__ _Float16 sH[16 * kE];     // 2 KB query rows f16
  __shared__ float sAtt[16 * kSEQP];   // 14 KB scores/attention, cols 200..223 = 0
  __shared__ float sRed[256];
  int b = blockIdx.x, t0 = blockIdx.y * 16;
  int lane = threadIdx.x & 31, wave = threadIdx.x >> 5;

  const float* sb = sup + (size_t)b * kSEQ * kE;
  for (int i = threadIdx.x; i < kSEQP * kE; i += 256)
    sS[i] = (i < kSEQ * kE) ? (_Float16)sb[i] : (_Float16)0.0f;
  const float* hrow = hA + ((size_t)b * kT + t0) * kE;
  for (int i = threadIdx.x; i < 16 * kE; i += 256) sH[i] = (_Float16)hrow[i];
  for (int i = threadIdx.x; i < 16 * kSEQP; i += 256) sAtt[i] = 0.0f;
  __syncthreads();

  // scores 16 x 208 (13 col tiles; cols 200..207 get 0 from zero B rows)
  for (int ct = wave; ct < 13; ct += 8) {
    v8f acc = {};
    acc = wmma16(fragA_lds16(sH, kE, 0, 0, lane),  fragBT_lds16(sS, kE, 0,  ct * 16, lane), acc);
    acc = wmma16(fragA_lds16(sH, kE, 0, 32, lane), fragBT_lds16(sS, kE, 32, ct * 16, lane), acc);
    storeC_ldsf32(sAtt, kSEQP, 0, ct * 16, acc, lane);
  }
  __syncthreads();

  // softmax over 200 cols (16 threads per row); pad cols stay 0 (= zero weight)
  int row = threadIdx.x >> 4, lc = threadIdx.x & 15;
  float m = -1e30f;
  for (int cix = lc; cix < kSEQ; cix += 16) m = fmaxf(m, sAtt[row * kSEQP + cix]);
  sRed[threadIdx.x] = m;
  __syncthreads();
  float rm = -1e30f;
  for (int i = 0; i < 16; ++i) rm = fmaxf(rm, sRed[row * 16 + i]);
  __syncthreads();
  float s = 0.0f;
  for (int cix = lc; cix < kSEQ; cix += 16) s += __expf(sAtt[row * kSEQP + cix] - rm);
  sRed[threadIdx.x] = s;
  __syncthreads();
  float rs = 0.0f;
  for (int i = 0; i < 16; ++i) rs += sRed[row * 16 + i];
  float inv = 1.0f / rs;
  for (int cix = lc; cix < kSEQ; cix += 16)
    sAtt[row * kSEQP + cix] = __expf(sAtt[row * kSEQP + cix] - rm) * inv;
  __syncthreads();

  // r = att(16x224) @ sup(224x64): 4 tiles on waves 0..3 (pad K contributes 0)
  if (wave < 4) {
    v8f acc = {};
    #pragma unroll
    for (int k0 = 0; k0 < kSEQP; k0 += 32) {
      v16h a  = fragA_ldsf32(sAtt, kSEQP, 0, k0, lane);
      v16h bb = fragB_lds16(sS, kE, k0, wave * 16, lane);
      acc = wmma16(a, bb, acc);
    }
    storeC_g(rout, kE, b * kT + t0, wave * 16, acc, lane);
  }
}

// Final cosine-sim softmax -> d_out (B,T,N,k) == (B,T,200) flat
__global__ void k_final(const float* __restrict__ tgt, const float* __restrict__ sup,
                        const float* __restrict__ sn, float* __restrict__ out) {
  __shared__ _Float16 sS[kSEQP * kE];
  __shared__ _Float16 sH[16 * kE];
  __shared__ float sAtt[16 * kSEQP];
  __shared__ float sRed[256];
  int b = blockIdx.x, t0 = blockIdx.y * 16;
  int lane = threadIdx.x & 31, wave = threadIdx.x >> 5;

  const float* sb = sup + (size_t)b * kSEQ * kE;
  for (int i = threadIdx.x; i < kSEQP * kE; i += 256)
    sS[i] = (i < kSEQ * kE) ? (_Float16)sb[i] : (_Float16)0.0f;
  const float* hrow = tgt + ((size_t)b * kT + t0) * kE;
  for (int i = threadIdx.x; i < 16 * kE; i += 256) sH[i] = (_Float16)hrow[i];
  for (int i = threadIdx.x; i < 16 * kSEQP; i += 256) sAtt[i] = 0.0f;
  __syncthreads();

  for (int ct = wave; ct < 13; ct += 8) {
    v8f acc = {};
    acc = wmma16(fragA_lds16(sH, kE, 0, 0, lane),  fragBT_lds16(sS, kE, 0,  ct * 16, lane), acc);
    acc = wmma16(fragA_lds16(sH, kE, 0, 32, lane), fragBT_lds16(sS, kE, 32, ct * 16, lane), acc);
    storeC_ldsf32(sAtt, kSEQP, 0, ct * 16, acc, lane);
  }
  __syncthreads();

  // target norms (f32 from global)
  int row = threadIdx.x >> 4, lc = threadIdx.x & 15;
  const float* trow = tgt + ((size_t)b * kT + t0 + row) * kE;
  float q = 0.0f;
  for (int e = lc; e < kE; e += 16) { float v = trow[e]; q += v * v; }
  sRed[threadIdx.x] = q;
  __syncthreads();
  float tn2 = 0.0f;
  for (int i = 0; i < 16; ++i) tn2 += sRed[row * 16 + i];
  float tn = sqrtf(tn2);
  __syncthreads();

  const float* snb = sn + (size_t)b * kSEQ;
  for (int cix = lc; cix < kSEQ; cix += 16) {
    float denom = fmaxf(tn * snb[cix], 1e-8f);
    sAtt[row * kSEQP + cix] = sAtt[row * kSEQP + cix] / denom;
  }
  __syncthreads();

  float m = -1e30f;
  for (int cix = lc; cix < kSEQ; cix += 16) m = fmaxf(m, sAtt[row * kSEQP + cix]);
  sRed[threadIdx.x] = m;
  __syncthreads();
  float rm = -1e30f;
  for (int i = 0; i < 16; ++i) rm = fmaxf(rm, sRed[row * 16 + i]);
  __syncthreads();
  float s = 0.0f;
  for (int cix = lc; cix < kSEQ; cix += 16) s += __expf(sAtt[row * kSEQP + cix] - rm);
  sRed[threadIdx.x] = s;
  __syncthreads();
  float rs = 0.0f;
  for (int i = 0; i < 16; ++i) rs += sRed[row * 16 + i];
  float inv = 1.0f / rs;
  float* orow = out + ((size_t)b * kT + t0 + row) * kSEQ;
  for (int cix = lc; cix < kSEQ; cix += 16)
    orow[cix] = __expf(sAtt[row * kSEQP + cix] - rm) * inv;
}

// =================== host launch ===================

extern "C" void kernel_launch(void* const* d_in, const int* in_sizes, int n_in,
                              void* d_out, int out_size, void* d_ws, size_t ws_size,
                              hipStream_t stream) {
  (void)in_sizes; (void)n_in; (void)out_size; (void)ws_size;
  const int*   sup_tok = (const int*)d_in[0];
  const int*   tgt_tok = (const int*)d_in[1];
  const float* emb     = (const float*)d_in[2];
  const float* f_Wih   = (const float*)d_in[3];
  const float* f_Whh   = (const float*)d_in[4];
  const float* f_bih   = (const float*)d_in[5];
  const float* f_bhh   = (const float*)d_in[6];
  const float* gf_Wih  = (const float*)d_in[7];
  const float* gf_Whh  = (const float*)d_in[8];
  const float* gf_bih  = (const float*)d_in[9];
  const float* gf_bhh  = (const float*)d_in[10];
  const float* gb_Wih  = (const float*)d_in[11];
  const float* gb_Whh  = (const float*)d_in[12];
  const float* gb_bih  = (const float*)d_in[13];
  const float* gb_bhh  = (const float*)d_in[14];

  float* p = (float*)d_ws;
  float* sup_enc = p; p += (size_t)kRS * kE;
  float* tgt_enc = p; p += (size_t)kRT * kE;
  float* xz_f    = p; p += (size_t)kRS * kG;
  float* xz_b    = p; p += (size_t)kRS * kG;
  float* hf      = p; p += (size_t)kRS * kE;
  float* hb      = p; p += (size_t)kRS * kE;
  float* semb    = p; p += (size_t)kRS * kE;
  float* sn      = p; p += (size_t)kRS;
  float* xz_t    = p; p += (size_t)kRT * kG;
  float* h0      = p; p += (size_t)kRT * kE;
  float* h1      = p; p += (size_t)kRT * kE;
  float* cbuf    = p; p += (size_t)kRT * kE;
  float* rbuf    = p; p += (size_t)kRT * kE;
  float* zbuf    = p; p += (size_t)kRT * kG;

  // encoders
  k_embed_sum<<<kRS, 64, 0, stream>>>(sup_tok, emb, sup_enc, kRS, kLS);
  k_embed_sum<<<kRT, 64, 0, stream>>>(tgt_tok, emb, tgt_enc, kRT, kLS);

  // BiLSTM: hoisted input GEMMs, then persistent recurrence
  k_gemm_xw<<<dim3(kRS / 16, 2), 256, 0, stream>>>(sup_enc, gf_Wih, gf_bih, gf_bhh, xz_f);
  k_gemm_xw<<<dim3(kRS / 16, 2), 256, 0, stream>>>(sup_enc, gb_Wih, gb_bih, gb_bhh, xz_b);
  k_bilstm<<<4, 512, 0, stream>>>(xz_f, xz_b, gf_Whh, gb_Whh, hf, hb);
  k_sup_combine<<<kRS, 64, 0, stream>>>(hf, hb, sup_enc, semb, sn);

  // target: hoisted input GEMM + state init
  k_gemm_xw<<<dim3(kRT / 16, 2), 256, 0, stream>>>(tgt_enc, f_Wih, f_bih, f_bhh, xz_t);
  k_zero3<<<(kRT * kE + 255) / 256, 256, 0, stream>>>(h0, cbuf, rbuf, kRT * kE);

  float* h = h0; float* hn = h1;
  for (int it = 0; it < 5; ++it) {
    k_add_inplace<<<(kRT * kE + 255) / 256, 256, 0, stream>>>(h, rbuf, kRT * kE);
    k_cell_z<<<dim3(kRT / 16, 2), 256, 0, stream>>>(h, f_Whh, xz_t, zbuf);
    k_gates<<<(kRT * kE + 255) / 256, 256, 0, stream>>>(zbuf, tgt_enc, cbuf, hn, kRT);
    k_att<<<dim3(kB, kT / 16), 256, 0, stream>>>(h, semb, rbuf);
    float* tmp = h; h = hn; hn = tmp;
  }

  k_final<<<dim3(kB, kT / 16), 256, 0, stream>>>(h, semb, sn, (float*)d_out);
}
